// ContinuousThoughtMachine_55611236548819
// MI455X (gfx1250) — compile-verified, matrix operand order checked by reading, not run on hardware
//
#include <hip/hip_runtime.h>
#include <hip/hip_bf16.h>
#include <math.h>

// ---------------------------------------------------------------------------
// Continuous Thought Machine for MI455X (gfx1250, wave32, WMMA).
// All GEMMs: v_wmma_f32_16x16x32_bf16 (bf16 in, fp32 accum).
// Each wave computes a 16x64 C strip: one A fragment feeds 4 WMMAs,
// quadrupling arithmetic intensity per A load.  Weights pre-transposed once
// to (N x K) bf16 so B-fragments are two contiguous 16B loads per lane;
// A (fp32 activations) converted to bf16 inline during fragment load.
// K/V cached in L2 as bf16 (67MB < 192MB L2).
// ---------------------------------------------------------------------------

typedef __bf16 bhalf;
typedef __attribute__((ext_vector_type(16))) __bf16 v16bf;
typedef __attribute__((ext_vector_type(8)))  __bf16 v8bf;
typedef __attribute__((ext_vector_type(8)))  float  v8f;

// problem constants (match reference)
constexpr int B_   = 32;
constexpr int S_   = 1024;
constexpr int DBB_ = 1024;
constexpr int E_   = 512;
constexpr int D_   = 2048;
constexpr int H_   = 8;      // heads, hd = 64
constexpr int MEM_ = 25;
constexpr int HID_ = 32;
constexpr int NSA_ = 512;
constexpr int NSO_ = 512;
constexpr int OUT_ = 1000;
constexpr int OUTP_= 1024;   // padded N for out GEMM
constexpr int IT_  = 50;

// ---------------------------------------------------------------------------
// block reductions (leading barrier protects LDS reuse)
// ---------------------------------------------------------------------------
__device__ __forceinline__ float blkSum(float* red, float v, int tid) {
  __syncthreads();
  red[tid] = v;
  __syncthreads();
  for (int s = 128; s > 0; s >>= 1) {
    if (tid < s) red[tid] += red[tid + s];
    __syncthreads();
  }
  return red[0];
}
__device__ __forceinline__ float blkMax(float* red, float v, int tid) {
  __syncthreads();
  red[tid] = v;
  __syncthreads();
  for (int s = 128; s > 0; s >>= 1) {
    if (tid < s) red[tid] = fmaxf(red[tid], red[tid + s]);
    __syncthreads();
  }
  return red[0];
}
__device__ __forceinline__ float sigm(float x) { return 1.f / (1.f + __expf(-x)); }

// ---------------------------------------------------------------------------
// WMMA GEMM: C(MxN) = A(MxK, fp32 row-major) * Bt(NxK, bf16)^T + bias
// One wave per 16x64 C strip (4 WMMAs per K step share one A fragment).
// K multiple of 32, N multiple of 64.  Epilogues:
//   Cperm != null : write bf16 permuted (b,h,s,d) for attention K/V
//   else          : write fp32 C with leading dim ldc
// A fragment (ISA 7.12.2, 16-bit A 16x32): lane l holds row (l&15),
// K chunks [kb..kb+7] and [16+kb..16+kb+7], kb = (l>>4)*8.
// B fragment = Bt row (col of B) with identical K packing.
// C fragment: lane l holds col (l&15); VGPR r holds row r + 8*(l>>4).
// ---------------------------------------------------------------------------
__global__ void gemm_bf16wmma(const float* __restrict__ A, int lda,
                              const bhalf* __restrict__ Bt, int K,
                              const float* __restrict__ bias, int biasN,
                              float* __restrict__ C, int ldc,
                              bhalf* __restrict__ Cperm,
                              int tilesN64, int totalTiles)
{
  int wave = (blockIdx.x * blockDim.x + threadIdx.x) >> 5;
  if (wave >= totalTiles) return;               // wave-uniform: EXEC stays all-1
  int lane = threadIdx.x & 31;
  int tm = wave / tilesN64;
  int tg = wave - tm * tilesN64;                // 64-wide N group
  int r16 = lane & 15;
  int kb  = (lane >> 4) << 3;

  const float* ap  = A  + (size_t)(tm * 16 + r16) * lda + kb;
  const bhalf* bp0 = Bt + (size_t)(tg * 64 + r16) * K   + kb;
  const size_t bstep = (size_t)16 * K;          // next 16-col sub-tile of B

  v8f c0, c1, c2, c3;
#pragma unroll
  for (int i = 0; i < 8; ++i) { c0[i] = 0.f; c1[i] = 0.f; c2[i] = 0.f; c3[i] = 0.f; }

  for (int kk = 0; kk < K; kk += 32) {
    if (kk + 32 < K) {                          // gfx1250 global_prefetch_b8
      __builtin_prefetch(ap + kk + 32, 0, 1);
      __builtin_prefetch(bp0 + kk + 32, 0, 1);
    }
    const float4 x0 = *(const float4*)(ap + kk);
    const float4 x1 = *(const float4*)(ap + kk + 4);
    const float4 x2 = *(const float4*)(ap + kk + 16);
    const float4 x3 = *(const float4*)(ap + kk + 20);
    v16bf a;
    a[0]=(bhalf)x0.x; a[1]=(bhalf)x0.y; a[2]=(bhalf)x0.z; a[3]=(bhalf)x0.w;
    a[4]=(bhalf)x1.x; a[5]=(bhalf)x1.y; a[6]=(bhalf)x1.z; a[7]=(bhalf)x1.w;
    a[8]=(bhalf)x2.x; a[9]=(bhalf)x2.y; a[10]=(bhalf)x2.z; a[11]=(bhalf)x2.w;
    a[12]=(bhalf)x3.x; a[13]=(bhalf)x3.y; a[14]=(bhalf)x3.z; a[15]=(bhalf)x3.w;

#pragma unroll
    for (int u = 0; u < 4; ++u) {
      const bhalf* bp = bp0 + (size_t)u * bstep + kk;
      const v8bf b0 = *(const v8bf*)(bp);
      const v8bf b1 = *(const v8bf*)(bp + 16);
      v16bf b;
#pragma unroll
      for (int e = 0; e < 8; ++e) { b[e] = b0[e]; b[8 + e] = b1[e]; }
      if      (u == 0) c0 = __builtin_amdgcn_wmma_f32_16x16x32_bf16(false, a, false, b, (short)0, c0, false, false);
      else if (u == 1) c1 = __builtin_amdgcn_wmma_f32_16x16x32_bf16(false, a, false, b, (short)0, c1, false, false);
      else if (u == 2) c2 = __builtin_amdgcn_wmma_f32_16x16x32_bf16(false, a, false, b, (short)0, c2, false, false);
      else             c3 = __builtin_amdgcn_wmma_f32_16x16x32_bf16(false, a, false, b, (short)0, c3, false, false);
    }
  }

  int mb = (tm << 4) + ((lane >> 4) << 3);
#pragma unroll
  for (int u = 0; u < 4; ++u) {
    const v8f& c = (u == 0) ? c0 : (u == 1) ? c1 : (u == 2) ? c2 : c3;
    int n = tg * 64 + u * 16 + (lane & 15);
    float bvv = (bias != nullptr && n < biasN) ? bias[n] : 0.f;
    if (Cperm) {
      // m = b*S + s ; n = h*64 + d  ->  out[((b*H + h)*S + s)*64 + d]
#pragma unroll
      for (int r = 0; r < 8; ++r) {
        int m = mb + r;
        int bb = m >> 10, s = m & 1023, h = n >> 6, d = n & 63;
        size_t idx = ((((size_t)(bb * H_ + h)) << 10) | (size_t)s);
        Cperm[(idx << 6) + d] = (bhalf)(c[r] + bvv);
      }
    } else {
#pragma unroll
      for (int r = 0; r < 8; ++r)
        C[(size_t)(mb + r) * ldc + n] = c[r] + bvv;
    }
  }
}

// ---------------------------------------------------------------------------
// one-time weight prep: Bt[n*K + k] = bf16(W[k*N + n]), zero-pad n >= N
// ---------------------------------------------------------------------------
__global__ void prep_weight_bf16t(const float* __restrict__ W, bhalf* __restrict__ Bt,
                                  int K, int N, int Np)
{
  int j = blockIdx.x * 256 + threadIdx.x;
  if (j >= Np * K) return;
  int n = j / K, k = j - n * K;
  float v = (n < N) ? W[(size_t)k * N + n] : 0.f;
  Bt[j] = (bhalf)v;
}

// nlm weight collapse: out[r] = sum_g W[r*D + g]   (einsum sums over g)
__global__ void nlm_reduce_g(const float* __restrict__ W, float* __restrict__ outp, int rows)
{
  int j = blockIdx.x * 256 + threadIdx.x;
  if (j >= rows) return;
  const float* p = W + (size_t)j * D_;
  float s = 0.f;
  for (int g = 0; g < D_; ++g) s += p[g];
  outp[j] = s;
}

// in-place LayerNorm over rows of width 512 (kv features)
__global__ void ln512_inplace(float* __restrict__ X,
                              const float* __restrict__ g, const float* __restrict__ bta)
{
  int row = blockIdx.x, tid = threadIdx.x;
  float* p = X + (size_t)row * 512;
  __shared__ float red[256];
  float x0 = p[tid], x1 = p[tid + 256];
  float m = blkSum(red, x0 + x1, tid) * (1.f / 512.f);
  float d0 = x0 - m, d1 = x1 - m;
  float var = blkSum(red, d0 * d0 + d1 * d1, tid) * (1.f / 512.f);
  float rs = rsqrtf(var + 1e-5f);
  p[tid]       = d0 * rs * g[tid]       + bta[tid];
  p[tid + 256] = d1 * rs * g[tid + 256] + bta[tid + 256];
}

// ---------------------------------------------------------------------------
// initial carry state
// ---------------------------------------------------------------------------
__global__ void init_state(const float* __restrict__ start_act,
                           const float* __restrict__ start_trace,
                           const int* __restrict__ il_o, const int* __restrict__ ir_o,
                           float* __restrict__ act, float* __restrict__ pre,
                           float* __restrict__ trace,
                           float* __restrict__ aa, float* __restrict__ ba,
                           float* __restrict__ ao, float* __restrict__ bo_)
{
  int j = blockIdx.x * 256 + threadIdx.x;
  if (j < B_ * D_) {
    int b = j >> 11, n = j & (D_ - 1);
    float a0 = start_act[n];
    act[j] = a0;
    pre[(size_t)b * (E_ + D_) + E_ + n] = a0;
    const float* st = start_trace + (size_t)n * MEM_;
    float* tr = trace + (size_t)j * MEM_;
    for (int i = 0; i < MEM_; ++i) tr[i] = st[i];
  }
  if (j < B_ * NSA_) { aa[j] = 0.f; ba[j] = 0.f; }
  if (j < B_ * NSO_) {
    int k = j & (NSO_ - 1);
    float p0 = start_act[il_o[k]] * start_act[ir_o[k]];
    ao[j] = p0; bo_[j] = 1.f;
  }
}

// ---------------------------------------------------------------------------
// action / out synchronization with exponential decay
// ---------------------------------------------------------------------------
__global__ void sync_action(const float* __restrict__ act,
                            const int* __restrict__ il, const int* __restrict__ ir,
                            const float* __restrict__ decay,
                            float* __restrict__ aa, float* __restrict__ ba,
                            float* __restrict__ syn_out, int iter)
{
  int j = blockIdx.x * 256 + threadIdx.x;
  if (j >= B_ * NSA_) return;
  int b = j >> 9, k = j & 511;
  float pa = act[(size_t)b * D_ + il[k]] * act[(size_t)b * D_ + ir[k]];
  float r  = __expf(-fminf(fmaxf(decay[k], 0.f), 15.f));
  float A, Bv;
  if (iter == 0) { A = pa; Bv = 1.f; }
  else           { A = r * aa[j] + pa; Bv = r * ba[j] + 1.f; }
  aa[j] = A; ba[j] = Bv;
  syn_out[j] = A * rsqrtf(Bv);
}

__global__ void sync_out_k(const float* __restrict__ act,
                           const int* __restrict__ il, const int* __restrict__ ir,
                           const float* __restrict__ decay,
                           float* __restrict__ ao, float* __restrict__ bo_,
                           float* __restrict__ syn_out, float* __restrict__ lastOut,
                           int isLast)
{
  int j = blockIdx.x * 256 + threadIdx.x;
  if (j >= B_ * NSO_) return;
  int b = j >> 9, k = j & 511;
  float po = act[(size_t)b * D_ + il[k]] * act[(size_t)b * D_ + ir[k]];
  float r  = __expf(-fminf(fmaxf(decay[k], 0.f), 15.f));
  float A  = r * ao[j]  + po;
  float Bv = r * bo_[j] + 1.f;
  ao[j] = A; bo_[j] = Bv;
  float s = A * rsqrtf(Bv);
  syn_out[j] = s;
  if (isLast) lastOut[j] = s;
}

// ---------------------------------------------------------------------------
// single-query attention per (b,h): scores over S=1024, softmax, V-weighted sum
// ---------------------------------------------------------------------------
__global__ void attention(const float* __restrict__ qh,
                          const bhalf* __restrict__ kh, const bhalf* __restrict__ vh,
                          float* __restrict__ attnH)
{
  int bh = blockIdx.x, tid = threadIdx.x;
  __shared__ float sQ[64];
  __shared__ float sSc[1024];
  __shared__ float red[256];

  if (tid < 64) sQ[tid] = qh[(size_t)bh * 64 + tid];
  __syncthreads();

  for (int s = tid; s < 1024; s += 256) {
    const bhalf* kp = kh + ((((size_t)bh) << 10) + s) * 64;
    float acc = 0.f;
#pragma unroll
    for (int d = 0; d < 64; ++d) acc += sQ[d] * (float)kp[d];
    sSc[s] = acc * 0.125f;                     // 1/sqrt(64)
  }

  float m = -3.4e38f;
  __syncthreads();
  for (int s = tid; s < 1024; s += 256) m = fmaxf(m, sSc[s]);
  m = blkMax(red, m, tid);

  float sum = 0.f;
  for (int s = tid; s < 1024; s += 256) {
    float e = __expf(sSc[s] - m);
    sSc[s] = e; sum += e;
  }
  sum = blkSum(red, sum, tid);                 // barriers make sSc visible
  float inv = 1.f / sum;

  int d = tid & 63, qg = tid >> 6;
  float acc = 0.f;
  const bhalf* vbase = vh + ((((size_t)bh) << 10) + (size_t)qg * 256) * 64 + d;
  const float* pbase = &sSc[qg * 256];
  for (int s = 0; s < 256; ++s) acc += pbase[s] * (float)vbase[(size_t)s * 64];

  __syncthreads();
  red[tid] = acc;
  __syncthreads();
  if (tid < 64) {
    float r = red[tid] + red[64 + tid] + red[128 + tid] + red[192 + tid];
    attnH[(size_t)bh * 64 + tid] = r * inv;
  }
}

// ---------------------------------------------------------------------------
// synapse epilogue: GLU over 4096 -> 2048, then LayerNorm over 2048
// ---------------------------------------------------------------------------
__global__ void glu_ln_synapse(const float* __restrict__ syn,
                               const float* __restrict__ g, const float* __restrict__ bta,
                               float* __restrict__ state)
{
  int b = blockIdx.x, tid = threadIdx.x;
  __shared__ float buf[D_];
  __shared__ float red[256];
  const float* p = syn + (size_t)b * (2 * D_);
  float psum = 0.f;
  for (int j = tid; j < D_; j += 256) {
    float v = p[j] * sigm(p[j + D_]);
    buf[j] = v; psum += v;
  }
  float m = blkSum(red, psum, tid) * (1.f / (float)D_);
  float pvar = 0.f;
  for (int j = tid; j < D_; j += 256) { float dd = buf[j] - m; pvar += dd * dd; }
  float var = blkSum(red, pvar, tid) * (1.f / (float)D_);
  float rs = rsqrtf(var + 1e-5f);
  for (int j = tid; j < D_; j += 256)
    state[(size_t)b * D_ + j] = (buf[j] - m) * rs * g[j] + bta[j];
}

// ---------------------------------------------------------------------------
// memory-trace shift + per-neuron NLMs (collapsed weights w1e 25x64, w2e 32x2)
// writes new act (fp32) and its copy into pre[:, 512:] for next step
// ---------------------------------------------------------------------------
__global__ void trace_nlm(float* __restrict__ trace, const float* __restrict__ state,
                          const float* __restrict__ w1e, const float* __restrict__ w2e,
                          const float* __restrict__ b1, const float* __restrict__ t1p,
                          const float* __restrict__ b2, const float* __restrict__ t2p,
                          float* __restrict__ act, float* __restrict__ pre)
{
  __shared__ float sw1[MEM_ * 64];
  __shared__ float sw2[HID_ * 2];
  int tid = threadIdx.x;
  for (int j = tid; j < MEM_ * 64; j += 256) sw1[j] = w1e[j];
  if (tid < HID_ * 2) sw2[tid] = w2e[tid];
  __syncthreads();

  int j = blockIdx.x * 256 + tid;
  if (j >= B_ * D_) return;
  int b = j >> 11, n = j & (D_ - 1);
  float t1 = t1p[0], t2 = t2p[0];

  float* trp = trace + (size_t)j * MEM_;
  float tr[MEM_];
#pragma unroll
  for (int i = 0; i < MEM_ - 1; ++i) tr[i] = trp[i + 1];
  tr[MEM_ - 1] = state[j];
#pragma unroll
  for (int i = 0; i < MEM_; ++i) trp[i] = tr[i];

  const float* b1p = b1 + (size_t)n * 64;
  float h[HID_];
  for (int o = 0; o < HID_; ++o) {
    float da = 0.f, db = 0.f;
#pragma unroll
    for (int i = 0; i < MEM_; ++i) {
      float t = tr[i];
      da += t * sw1[i * 64 + o];
      db += t * sw1[i * 64 + 32 + o];
    }
    float sa = (da + b1p[o]) * t1;
    float sb = (db + b1p[32 + o]) * t1;
    h[o] = sa * sigm(sb);
  }
  float y0 = 0.f, y1 = 0.f;
#pragma unroll
  for (int i = 0; i < HID_; ++i) { y0 += h[i] * sw2[i * 2]; y1 += h[i] * sw2[i * 2 + 1]; }
  y0 = (y0 + b2[(size_t)n * 2])     * t2;
  y1 = (y1 + b2[(size_t)n * 2 + 1]) * t2;
  float a = y0 * sigm(y1);
  act[j] = a;
  pre[(size_t)b * (E_ + D_) + E_ + n] = a;
}

// ---------------------------------------------------------------------------
// predictions + normalized-entropy certainties -> d_out (strided over ITERS)
// ---------------------------------------------------------------------------
__global__ void out_softent(const float* __restrict__ pred, float* __restrict__ dout, int iter)
{
  int b = blockIdx.x, tid = threadIdx.x;
  __shared__ float red[256];
  const float* p = pred + (size_t)b * OUTP_;
  float m = -3.4e38f;
  for (int o = tid; o < OUT_; o += 256) m = fmaxf(m, p[o]);
  m = blkMax(red, m, tid);
  float s = 0.f;
  for (int o = tid; o < OUT_; o += 256) s += __expf(p[o] - m);
  s = blkSum(red, s, tid);
  float logZ = m + __logf(s);
  float ent = 0.f;
  for (int o = tid; o < OUT_; o += 256) {
    float x = p[o];
    float lp = x - logZ;
    ent += __expf(lp) * lp;
    dout[((size_t)b * OUT_ + o) * IT_ + iter] = x;
  }
  ent = blkSum(red, ent, tid);
  if (tid == 0) {
    float ne = -ent * (1.f / 6.90775527898f);  // / log(1000)
    size_t cb = (size_t)B_ * OUT_ * IT_;
    dout[cb + ((size_t)b * 2 + 0) * IT_ + iter] = ne;
    dout[cb + ((size_t)b * 2 + 1) * IT_ + iter] = 1.f - ne;
  }
}

// ---------------------------------------------------------------------------
// host launcher
// ---------------------------------------------------------------------------
extern "C" void kernel_launch(void* const* d_in, const int* in_sizes, int n_in,
                              void* d_out, int out_size, void* d_ws, size_t ws_size,
                              hipStream_t stream)
{
  (void)in_sizes; (void)n_in; (void)out_size; (void)ws_size;
  const float* x      = (const float*)d_in[0];
  const float* kv_w   = (const float*)d_in[1];
  const float* kv_b   = (const float*)d_in[2];
  const float* kv_g   = (const float*)d_in[3];
  const float* kv_bb  = (const float*)d_in[4];
  const float* q_w    = (const float*)d_in[5];
  const float* q_b    = (const float*)d_in[6];
  const float* wq     = (const float*)d_in[7];
  const float* bq     = (const float*)d_in[8];
  const float* wk     = (const float*)d_in[9];
  const float* bk     = (const float*)d_in[10];
  const float* wv     = (const float*)d_in[11];
  const float* bv     = (const float*)d_in[12];
  const float* wo     = (const float*)d_in[13];
  const float* bo     = (const float*)d_in[14];
  const float* syn_w  = (const float*)d_in[15];
  const float* syn_b  = (const float*)d_in[16];
  const float* syn_g  = (const float*)d_in[17];
  const float* syn_bb = (const float*)d_in[18];
  const float* nlm_w1 = (const float*)d_in[19];
  const float* nlm_b1 = (const float*)d_in[20];
  const float* nlm_t1 = (const float*)d_in[21];
  const float* nlm_w2 = (const float*)d_in[22];
  const float* nlm_b2 = (const float*)d_in[23];
  const float* nlm_t2 = (const float*)d_in[24];
  const float* start_trace = (const float*)d_in[25];
  const float* start_act   = (const float*)d_in[26];
  const float* decay_a = (const float*)d_in[27];
  const float* decay_o = (const float*)d_in[28];
  const float* out_w  = (const float*)d_in[29];
  const float* out_b  = (const float*)d_in[30];
  const int* il_a = (const int*)d_in[31];
  const int* ir_a = (const int*)d_in[32];
  const int* il_o = (const int*)d_in[33];
  const int* ir_o = (const int*)d_in[34];
  float* out = (float*)d_out;

  char* wsb = (char*)d_ws;
  size_t off = 0;
  auto walloc = [&](size_t bytes) -> void* {
    void* p = wsb + off;
    off += (bytes + 255) & ~(size_t)255;
    return p;
  };

  float* kvt    = (float*)walloc((size_t)B_ * S_ * E_ * 4);         // kv (LN in-place)
  bhalf* kh     = (bhalf*)walloc((size_t)B_ * S_ * E_ * 2);         // (b,h,s,d) bf16
  bhalf* vh     = (bhalf*)walloc((size_t)B_ * S_ * E_ * 2);
  bhalf* kv_wb  = (bhalf*)walloc((size_t)E_ * DBB_ * 2);
  bhalf* q_wb   = (bhalf*)walloc((size_t)E_ * NSA_ * 2);
  bhalf* wqb    = (bhalf*)walloc((size_t)E_ * E_ * 2);
  bhalf* wkb    = (bhalf*)walloc((size_t)E_ * E_ * 2);
  bhalf* wvb    = (bhalf*)walloc((size_t)E_ * E_ * 2);
  bhalf* wob    = (bhalf*)walloc((size_t)E_ * E_ * 2);
  bhalf* syn_wb = (bhalf*)walloc((size_t)(2 * D_) * (E_ + D_) * 2);
  bhalf* out_wb = (bhalf*)walloc((size_t)OUTP_ * E_ * 2);
  float* w1e    = (float*)walloc((size_t)MEM_ * 2 * HID_ * 4);
  float* w2e    = (float*)walloc((size_t)HID_ * 2 * 4);
  float* act    = (float*)walloc((size_t)B_ * D_ * 4);
  float* trace  = (float*)walloc((size_t)B_ * D_ * MEM_ * 4);
  float* aa     = (float*)walloc((size_t)B_ * NSA_ * 4);
  float* ba     = (float*)walloc((size_t)B_ * NSA_ * 4);
  float* ao     = (float*)walloc((size_t)B_ * NSO_ * 4);
  float* bo_    = (float*)walloc((size_t)B_ * NSO_ * 4);
  float* syncA  = (float*)walloc((size_t)B_ * NSA_ * 4);
  float* syncO  = (float*)walloc((size_t)B_ * NSO_ * 4);
  float* qfull  = (float*)walloc((size_t)B_ * E_ * 4);
  float* qhB    = (float*)walloc((size_t)B_ * E_ * 4);
  float* attnH  = (float*)walloc((size_t)B_ * E_ * 4);
  float* pre    = (float*)walloc((size_t)B_ * (E_ + D_) * 4);
  float* syn    = (float*)walloc((size_t)B_ * 2 * D_ * 4);
  float* state  = (float*)walloc((size_t)B_ * D_ * 4);
  float* pred   = (float*)walloc((size_t)B_ * OUTP_ * 4);

  auto g1 = [](int n) { return (n + 255) / 256; };
  auto gw = [](int waves) { return (waves * 32 + 255) / 256; };  // 8 waves/block

  // -------- one-time prep --------
  prep_weight_bf16t<<<g1(E_ * DBB_), 256, 0, stream>>>(kv_w, kv_wb, DBB_, E_, E_);
  prep_weight_bf16t<<<g1(E_ * NSA_), 256, 0, stream>>>(q_w,  q_wb,  NSA_, E_, E_);
  prep_weight_bf16t<<<g1(E_ * E_),   256, 0, stream>>>(wq,   wqb,   E_,   E_, E_);
  prep_weight_bf16t<<<g1(E_ * E_),   256, 0, stream>>>(wk,   wkb,   E_,   E_, E_);
  prep_weight_bf16t<<<g1(E_ * E_),   256, 0, stream>>>(wv,   wvb,   E_,   E_, E_);
  prep_weight_bf16t<<<g1(E_ * E_),   256, 0, stream>>>(wo,   wob,   E_,   E_, E_);
  prep_weight_bf16t<<<g1((2 * D_) * (E_ + D_)), 256, 0, stream>>>(syn_w, syn_wb, E_ + D_, 2 * D_, 2 * D_);
  prep_weight_bf16t<<<g1(OUTP_ * E_), 256, 0, stream>>>(out_w, out_wb, E_, OUT_, OUTP_);
  nlm_reduce_g<<<g1(MEM_ * 2 * HID_), 256, 0, stream>>>(nlm_w1, w1e, MEM_ * 2 * HID_);
  nlm_reduce_g<<<g1(HID_ * 2), 256, 0, stream>>>(nlm_w2, w2e, HID_ * 2);

  // kv = LN(x @ kv_w + kv_b) ; kh/vh = (kv @ wk/wv + b) permuted bf16
  {
    int tn = E_ / 64, tot = (B_ * S_ / 16) * tn;
    gemm_bf16wmma<<<gw(tot), 256, 0, stream>>>(x, DBB_, kv_wb, DBB_, kv_b, E_,
                                               kvt, E_, nullptr, tn, tot);
    ln512_inplace<<<B_ * S_, 256, 0, stream>>>(kvt, kv_g, kv_bb);
    gemm_bf16wmma<<<gw(tot), 256, 0, stream>>>(kvt, E_, wkb, E_, bk, E_,
                                               nullptr, 0, kh, tn, tot);
    gemm_bf16wmma<<<gw(tot), 256, 0, stream>>>(kvt, E_, wvb, E_, bv, E_,
                                               nullptr, 0, vh, tn, tot);
  }

  init_state<<<g1(B_ * D_), 256, 0, stream>>>(start_act, start_trace, il_o, ir_o,
                                              act, pre, trace, aa, ba, ao, bo_);

  float* syncLast = out + (size_t)B_ * OUT_ * IT_ + (size_t)B_ * 2 * IT_;

  // -------- 50-step scan --------
  for (int it = 0; it < IT_; ++it) {
    sync_action<<<g1(B_ * NSA_), 256, 0, stream>>>(act, il_a, ir_a, decay_a,
                                                   aa, ba, syncA, it);
    {
      int tn = E_ / 64, tot = (B_ / 16) * tn;
      gemm_bf16wmma<<<gw(tot), 256, 0, stream>>>(syncA, NSA_, q_wb, NSA_, q_b, E_,
                                                 qfull, E_, nullptr, tn, tot);
      gemm_bf16wmma<<<gw(tot), 256, 0, stream>>>(qfull, E_, wqb, E_, bq, E_,
                                                 qhB, E_, nullptr, tn, tot);
    }
    attention<<<B_ * H_, 256, 0, stream>>>(qhB, kh, vh, attnH);
    {
      int tn = E_ / 64, tot = (B_ / 16) * tn;
      gemm_bf16wmma<<<gw(tot), 256, 0, stream>>>(attnH, E_, wob, E_, bo, E_,
                                                 pre, E_ + D_, nullptr, tn, tot);
    }
    {
      int tn = (2 * D_) / 64, tot = (B_ / 16) * tn;
      gemm_bf16wmma<<<gw(tot), 256, 0, stream>>>(pre, E_ + D_, syn_wb, E_ + D_,
                                                 syn_b, 2 * D_, syn, 2 * D_,
                                                 nullptr, tn, tot);
    }
    glu_ln_synapse<<<B_, 256, 0, stream>>>(syn, syn_g, syn_bb, state);
    trace_nlm<<<g1(B_ * D_), 256, 0, stream>>>(trace, state, w1e, w2e,
                                               nlm_b1, nlm_t1, nlm_b2, nlm_t2,
                                               act, pre);
    sync_out_k<<<g1(B_ * NSO_), 256, 0, stream>>>(act, il_o, ir_o, decay_o,
                                                  ao, bo_, syncO, syncLast,
                                                  (it == IT_ - 1) ? 1 : 0);
    {
      int tn = OUTP_ / 64, tot = (B_ / 16) * tn;
      gemm_bf16wmma<<<gw(tot), 256, 0, stream>>>(syncO, NSO_, out_wb, NSO_,
                                                 out_b, OUT_, pred, OUTP_,
                                                 nullptr, tn, tot);
    }
    out_softent<<<B_, 256, 0, stream>>>(pred, out, it);
  }
}
